// GraniteAttention_68848325755259
// MI455X (gfx1250) — compile-verified, matrix operand order checked by reading, not run on hardware
//
#include <hip/hip_runtime.h>

// ---------------------------------------------------------------------------
// GQA attention layer for MI455X (gfx1250, wave32, WMMA bf16 16x16x32).
// B=1, S=2048, E=4096, H=32, KV=8, G=4, D=128.
// GEMMs: double-buffered LDS pipeline fed by GLOBAL_LOAD_ASYNC_TO_LDS_B128
// (ASYNCcnt-tracked), compute via v_wmma_f32_16x16x32_bf16.
// ---------------------------------------------------------------------------

#define S_LEN 2048
#define E_DIM 4096
#define H_NUM 32
#define KV_NUM 8
#define G_NUM 4
#define D_DIM 128
#define FQKV 6144                 // (G+2)*D*KV
#define ATTN_SCALE 0.08838834764831845f   // 1/sqrt(128)

#define QBLK 64
#define KBLK 32

typedef __attribute__((ext_vector_type(16))) __bf16 v16bf;
typedef __attribute__((ext_vector_type(8)))  float  v8f;
typedef __attribute__((ext_vector_type(4)))  unsigned int v4u;
typedef unsigned short u16;

union FragU  { v16bf v; v4u u[2]; };
union Pack16 { v4u u[2]; u16 s[16]; };

__device__ __forceinline__ u16 f32_bf16(float f) {
  unsigned int u = __float_as_uint(f);
  u += 0x7FFFu + ((u >> 16) & 1u);          // round-to-nearest-even
  return (u16)(u >> 16);
}

// Async DMA of 16 bytes global -> LDS (no VGPR staging; tracked by ASYNCcnt).
// LDS destination address = low 32 bits of the generic shared pointer.
__device__ __forceinline__ void async_copy_b128(void* lds, const void* g) {
  unsigned loff = (unsigned)(size_t)lds;
  unsigned long long ga = (unsigned long long)(size_t)g;
  asm volatile("global_load_async_to_lds_b128 %0, %1, off"
               :: "v"(loff), "v"(ga) : "memory");
}
__device__ __forceinline__ void wait_async() {
  asm volatile("s_wait_asynccnt 0x0" ::: "memory");
}

// Load one 16x32 bf16 WMMA A/B fragment from an LDS tile stored row-major.
// ISA layout: lanes 0-15 hold row (base+lane), K = {k0+0..7, k0+16..23};
// lanes 16-31 hold K = {k0+8..15, k0+24..31}. Two ds_load_b128 per lane.
__device__ __forceinline__ v16bf lds_frag(const u16* base, int row, int stride,
                                          int k0, int lane) {
  const u16* p = base + (size_t)row * stride + k0 + ((lane >> 4) << 3);
  FragU f;
  f.u[0] = *(const v4u*)p;
  f.u[1] = *(const v4u*)(p + 16);
  return f.v;
}

// ---------------------------------------------------------------------------
// f32 -> bf16 conversion (grid-stride)
// ---------------------------------------------------------------------------
__global__ void cvt_f32_bf16(const float* __restrict__ in, u16* __restrict__ out,
                             int n) {
  for (int i = blockIdx.x * blockDim.x + threadIdx.x; i < n;
       i += gridDim.x * blockDim.x)
    out[i] = f32_bf16(in[i]);
}

// ---------------------------------------------------------------------------
// C[M,N] f32 = A[M,K] (bf16, row-major) x B[N,K]^T (bf16, row-major rows = N)
// Block: 256 threads / 8 waves, 128x128 output tile, BK=32, double-buffered
// async-to-LDS staging. Wave (wm 0..3, wn 0..1) owns 32x64 = 2x4 WMMA tiles.
// ---------------------------------------------------------------------------
__launch_bounds__(256)
__global__ void gemm_bf16(const u16* __restrict__ A, const u16* __restrict__ B,
                          float* __restrict__ C, int M, int N, int K) {
  __shared__ u16 As[2][128][40];  // 32 cols + 8 skew -> conflict-free b128
  __shared__ u16 Bs[2][128][40];
  const int tid = threadIdx.x, wave = tid >> 5, lane = tid & 31;
  const int wm = wave >> 1, wn = wave & 1;
  const int bm = blockIdx.y, bn = blockIdx.x;
  const u16* Ab = A + (size_t)bm * 128 * K;
  const u16* Bb = B + (size_t)bn * 128 * K;
  const int lrow = tid >> 1, lseg = (tid & 1) * 16;  // 2 threads/row, 32B each
  const int rhi = (lane >> 4) * 8, cl = lane & 15;
  v8f acc[2][4] = {};

  // stage one 128x32 A tile + B tile into buffer `buf` (4 async b128 / thread)
  auto stage = [&](int buf, int k0) {
    const u16* ga = Ab + (size_t)lrow * K + k0 + lseg;
    const u16* gb = Bb + (size_t)lrow * K + k0 + lseg;
    async_copy_b128(&As[buf][lrow][lseg],     ga);
    async_copy_b128(&As[buf][lrow][lseg + 8], ga + 8);
    async_copy_b128(&Bs[buf][lrow][lseg],     gb);
    async_copy_b128(&Bs[buf][lrow][lseg + 8], gb + 8);
  };

  stage(0, 0);
  wait_async();
  __syncthreads();

  for (int k0 = 0; k0 < K; k0 += 32) {
    const int buf = (k0 >> 5) & 1;
    if (k0 + 32 < K) stage(buf ^ 1, k0 + 32);   // prefetch next tile (async DMA)

    v16bf af[2], bfv[4];
#pragma unroll
    for (int i = 0; i < 2; i++)
      af[i] = lds_frag(&As[buf][0][0], wm * 32 + i * 16 + cl, 40, 0, lane);
#pragma unroll
    for (int j = 0; j < 4; j++)
      bfv[j] = lds_frag(&Bs[buf][0][0], wn * 64 + j * 16 + cl, 40, 0, lane);
#pragma unroll
    for (int i = 0; i < 2; i++)
#pragma unroll
      for (int j = 0; j < 4; j++)
        acc[i][j] = __builtin_amdgcn_wmma_f32_16x16x32_bf16(
            false, af[i], false, bfv[j], (short)0, acc[i][j], false, false);

    wait_async();     // own async writes to buf^1 done
    __syncthreads();  // everyone finished reading buf + wrote buf^1
  }

#pragma unroll
  for (int i = 0; i < 2; i++)
#pragma unroll
    for (int j = 0; j < 4; j++)
#pragma unroll
      for (int r = 0; r < 8; r++) {
        int row = bm * 128 + wm * 32 + i * 16 + r + rhi;
        int col = bn * 128 + wn * 64 + j * 16 + cl;
        C[(size_t)row * N + col] = acc[i][j][r];
      }
}

// ---------------------------------------------------------------------------
// RoPE + GQA reorder: qkv f32 [S, KV, (G+2)*D] -> Q bf16 [H,S,D],
// K bf16 [KV,S,D] (both roped), V bf16 [KV,S,D].
// ---------------------------------------------------------------------------
__global__ void rope_reorder(const float* __restrict__ qkv,
                             const float* __restrict__ cb,
                             const float* __restrict__ sb,
                             u16* __restrict__ Qo, u16* __restrict__ Ko,
                             u16* __restrict__ Vo) {
  const size_t NQ = (size_t)H_NUM * S_LEN * D_DIM;
  const size_t NK = (size_t)KV_NUM * S_LEN * D_DIM;
  size_t idx = (size_t)blockIdx.x * blockDim.x + threadIdx.x;
  if (idx < NQ) {
    int h = (int)(idx / (S_LEN * D_DIM));
    int rem = (int)(idx % (S_LEN * D_DIM));
    int s = rem / D_DIM, d = rem % D_DIM;
    int kv = h / G_NUM, g = h % G_NUM;
    const float* row = qkv + (size_t)s * FQKV + kv * ((G_NUM + 2) * D_DIM) + g * D_DIM;
    float x = row[d];
    float rot = (d < 64) ? -row[d + 64] : row[d - 64];
    float c = cb[(size_t)s * D_DIM + d], sn = sb[(size_t)s * D_DIM + d];
    Qo[idx] = f32_bf16(x * c + rot * sn);
  } else if (idx < NQ + NK) {
    size_t i2 = idx - NQ;
    int kv = (int)(i2 / (S_LEN * D_DIM));
    int rem = (int)(i2 % (S_LEN * D_DIM));
    int s = rem / D_DIM, d = rem % D_DIM;
    const float* row = qkv + (size_t)s * FQKV + kv * ((G_NUM + 2) * D_DIM) + G_NUM * D_DIM;
    float x = row[d];
    float rot = (d < 64) ? -row[d + 64] : row[d - 64];
    float c = cb[(size_t)s * D_DIM + d], sn = sb[(size_t)s * D_DIM + d];
    Ko[i2] = f32_bf16(x * c + rot * sn);
  } else if (idx < NQ + 2 * NK) {
    size_t i2 = idx - NQ - NK;
    int kv = (int)(i2 / (S_LEN * D_DIM));
    int rem = (int)(i2 % (S_LEN * D_DIM));
    int s = rem / D_DIM, d = rem % D_DIM;
    Vo[i2] = f32_bf16(
        qkv[(size_t)s * FQKV + kv * ((G_NUM + 2) * D_DIM) + (G_NUM + 1) * D_DIM + d]);
  }
}

// ---------------------------------------------------------------------------
// Flash attention, causal. One block per (head, 64-query tile); 8 waves.
// Q and K tiles staged via async-to-LDS DMA; V staged transposed (manual
// scatter) so its WMMA B-fragment is a row-gather. Online softmax in LDS;
// running O kept in registers (4 x v8f per wave).
// ---------------------------------------------------------------------------
__launch_bounds__(256)
__global__ void attn_fwd(const u16* __restrict__ Qg, const u16* __restrict__ Kg,
                         const u16* __restrict__ Vg, u16* __restrict__ ctx) {
  __shared__ u16 Qs[QBLK][136];       // 64 x 128 bf16 (+8 skew)
  __shared__ u16 Ks[KBLK][136];       // 32 x 128 bf16
  __shared__ u16 Vt[D_DIM][40];       // V transposed: [d][key]
  __shared__ float Sc[QBLK][36];      // raw scores 64 x 32
  __shared__ u16 Ps[QBLK][40];        // softmax numerators bf16
  __shared__ float mrow[QBLK], lrow[QBLK], srow[QBLK];

  const int h = blockIdx.x, qb = blockIdx.y;
  const int kvh = h / G_NUM;
  const int tid = threadIdx.x, wave = tid >> 5, lane = tid & 31;
  const int wm = wave >> 1, wn = wave & 1;
  const int rhi = (lane >> 4) * 8, cl = lane & 15;
  const u16* Qh = Qg + ((size_t)h * S_LEN + qb * QBLK) * D_DIM;
  const u16* Kh = Kg + (size_t)kvh * S_LEN * D_DIM;
  const u16* Vh = Vg + (size_t)kvh * S_LEN * D_DIM;

  { // async-stage Q tile: each thread owns 32 bf16 (4 x b128)
    int row = tid >> 2, seg = (tid & 3) * 32;
    const u16* src = Qh + (size_t)row * D_DIM + seg;
    async_copy_b128(&Qs[row][seg],      src);
    async_copy_b128(&Qs[row][seg + 8],  src + 8);
    async_copy_b128(&Qs[row][seg + 16], src + 16);
    async_copy_b128(&Qs[row][seg + 24], src + 24);
  }
  if (tid < QBLK) { mrow[tid] = -1e30f; lrow[tid] = 0.f; }

  v8f acc[4] = {};
  const int kbmax = 2 * qb + 1;       // causal: key block start <= last q row

  for (int kb = 0; kb <= kbmax; ++kb) {
    __syncthreads();                  // prior P*V done before K/V/Ps reuse
    { // stage K (async DMA) and V transposed (manual scatter)
      int row = tid >> 3, seg = (tid & 7) * 16;
      const u16* sk = Kh + ((size_t)(kb * KBLK + row)) * D_DIM + seg;
      async_copy_b128(&Ks[row][seg],     sk);
      async_copy_b128(&Ks[row][seg + 8], sk + 8);
      Pack16 pv;
      const v4u* svp = (const v4u*)(Vh + ((size_t)(kb * KBLK + row)) * D_DIM + seg);
      pv.u[0] = svp[0]; pv.u[1] = svp[1];
#pragma unroll
      for (int e = 0; e < 16; e++) Vt[seg + e][row] = pv.s[e];
    }
    wait_async();                     // Q (first iter) + K DMA visible in LDS
    __syncthreads();

    // --- scores: each wave one 16x16 tile of the 64x32 score block ---
    v8f sco = {};
#pragma unroll
    for (int kk = 0; kk < 4; kk++) {
      v16bf aq = lds_frag(&Qs[0][0], wm * 16 + cl, 136, kk * 32, lane);
      v16bf bk = lds_frag(&Ks[0][0], wn * 16 + cl, 136, kk * 32, lane);
      sco = __builtin_amdgcn_wmma_f32_16x16x32_bf16(false, aq, false, bk,
                                                    (short)0, sco, false, false);
    }
#pragma unroll
    for (int r = 0; r < 8; r++) {
      int row = wm * 16 + r + rhi;
      int col = wn * 16 + cl;
      int qg = qb * QBLK + row, kg = kb * KBLK + col;
      float x = sco[r] * ATTN_SCALE;
      if (kg > qg) x = -1e30f;        // additive causal mask
      Sc[row][col] = x;
    }
    __syncthreads();

    // --- online softmax: one thread per query row ---
    if (tid < QBLK) {
      float mo = mrow[tid];
      float mb = mo;
#pragma unroll 4
      for (int c = 0; c < KBLK; c++) mb = fmaxf(mb, Sc[tid][c]);
      float scl = __expf(mo - mb);
      float lsum = 0.f;
#pragma unroll 4
      for (int c = 0; c < KBLK; c++) {
        float xv = Sc[tid][c];
        float e = (xv > -1e29f) ? __expf(xv - mb) : 0.f;
        lsum += e;
        Ps[tid][c] = f32_bf16(e);
      }
      mrow[tid] = mb;
      lrow[tid] = lrow[tid] * scl + lsum;
      srow[tid] = scl;
    }
    __syncthreads();

    // --- rescale running O, then O += P x V (wave owns 16 rows x 64 cols) ---
    float fac[8];
#pragma unroll
    for (int r = 0; r < 8; r++) fac[r] = srow[wm * 16 + r + rhi];
    v16bf ap = lds_frag(&Ps[0][0], wm * 16 + cl, 40, 0, lane);
#pragma unroll
    for (int j = 0; j < 4; j++) {
#pragma unroll
      for (int r = 0; r < 8; r++) acc[j][r] *= fac[r];
      v16bf bv = lds_frag(&Vt[0][0], wn * 64 + j * 16 + cl, 40, 0, lane);
      acc[j] = __builtin_amdgcn_wmma_f32_16x16x32_bf16(false, ap, false, bv,
                                                       (short)0, acc[j], false, false);
    }
  }

  // --- epilogue: divide by l, write context bf16 [S, H*D] ---
#pragma unroll
  for (int j = 0; j < 4; j++)
#pragma unroll
    for (int r = 0; r < 8; r++) {
      int row = wm * 16 + r + rhi;
      float invl = 1.f / lrow[row];
      int sg = qb * QBLK + row;
      int dg = h * D_DIM + wn * 64 + j * 16 + cl;
      ctx[(size_t)sg * (H_NUM * D_DIM) + dg] = f32_bf16(acc[j][r] * invl);
    }
}

// ---------------------------------------------------------------------------
// Host-side orchestration (graph-capture safe: launches only, ws for scratch)
// ---------------------------------------------------------------------------
extern "C" void kernel_launch(void* const* d_in, const int* in_sizes, int n_in,
                              void* d_out, int out_size, void* d_ws, size_t ws_size,
                              hipStream_t stream) {
  const float* hidden = (const float*)d_in[0];
  // d_in[1] = attention_mask (pure causal -> applied analytically in-kernel)
  const float* cosb   = (const float*)d_in[2];
  const float* sinb   = (const float*)d_in[3];
  const float* wqkv   = (const float*)d_in[4];
  const float* wproj  = (const float*)d_in[5];
  float* out = (float*)d_out;

  char* ws = (char*)d_ws;
  size_t off = 0;
  auto take = [&](size_t bytes) {
    char* p = ws + off;
    off = (off + bytes + 255) & ~(size_t)255;
    return p;
  };
  u16*   hb   = (u16*)  take((size_t)S_LEN * E_DIM * 2);   // hidden bf16
  u16*   wqb  = (u16*)  take((size_t)FQKV * E_DIM * 2);    // w_qkv bf16 (L2-resident)
  u16*   wpb  = (u16*)  take((size_t)E_DIM * E_DIM * 2);   // w_proj bf16
  float* qkvf = (float*)take((size_t)S_LEN * FQKV * 4);    // qkv f32
  u16*   Qb   = (u16*)  take((size_t)H_NUM * S_LEN * D_DIM * 2);
  u16*   Kb   = (u16*)  take((size_t)KV_NUM * S_LEN * D_DIM * 2);
  u16*   Vb   = (u16*)  take((size_t)KV_NUM * S_LEN * D_DIM * 2);
  u16*   ctx  = (u16*)  take((size_t)S_LEN * E_DIM * 2);

  cvt_f32_bf16<<<2048, 256, 0, stream>>>(hidden, hb, S_LEN * E_DIM);
  cvt_f32_bf16<<<4096, 256, 0, stream>>>(wqkv, wqb, FQKV * E_DIM);
  cvt_f32_bf16<<<4096, 256, 0, stream>>>(wproj, wpb, E_DIM * E_DIM);

  gemm_bf16<<<dim3(FQKV / 128, S_LEN / 128), 256, 0, stream>>>(
      hb, wqb, qkvf, S_LEN, FQKV, E_DIM);

  size_t tot = (size_t)(H_NUM + 2 * KV_NUM) * S_LEN * D_DIM;
  rope_reorder<<<(unsigned)((tot + 255) / 256), 256, 0, stream>>>(
      qkvf, cosb, sinb, Qb, Kb, Vb);

  attn_fwd<<<dim3(H_NUM, S_LEN / QBLK), 256, 0, stream>>>(Qb, Kb, Vb, ctx);

  gemm_bf16<<<dim3(E_DIM / 128, S_LEN / 128), 256, 0, stream>>>(
      ctx, wpb, out, S_LEN, E_DIM, E_DIM);

  (void)in_sizes; (void)n_in; (void)out_size; (void)ws_size;
}
